// SelfAttention_44976897524184
// MI455X (gfx1250) — matrix operand order repaired and checked by reading, not compiled, not run on hardware
//
#include <hip/hip_runtime.h>
#include <hip/hip_bf16.h>

typedef __attribute__((ext_vector_type(16))) __bf16 v16bf;
typedef __attribute__((ext_vector_type(8)))  __bf16 v8bf;
typedef __attribute__((ext_vector_type(8)))  float  v8f;
typedef __attribute__((address_space(3)))    __bf16 as3_bf16;

#define DEVFN __device__ __forceinline__

constexpr int BATCH = 2;
constexpr int C     = 128;     // channels / head dim
constexpr int N     = 4096;    // 16*16*16 voxels
constexpr int G     = 8;       // split-K groups over the key dimension
constexpr int NT    = N / 16;  // 256 query tiles per batch
constexpr int JPG   = N / 64 / G;  // 64-key tiles per group = 8

// ---------------------------------------------------------------------------
// Async global->LDS copy, 16B per lane (ASYNCcnt path).  VGLOBAL encoding:
// VDST = 32-bit LDS byte address, VADDR = 64-bit global address.
// ---------------------------------------------------------------------------
DEVFN void async_cp16(const __bf16* g, __bf16* l) {
  unsigned lds_addr = (unsigned)(unsigned long long)(as3_bf16*)l;
  asm volatile("global_load_async_to_lds_b128 %0, %1, off"
               :
               : "v"(lds_addr), "v"(g)
               : "memory");
}

DEVFN void wait_async() {
#if __has_builtin(__builtin_amdgcn_s_wait_asynccnt)
  __builtin_amdgcn_s_wait_asynccnt(0);
#else
  asm volatile("s_wait_asynccnt 0x0" ::: "memory");
#endif
}

// ---------------------------------------------------------------------------
// WMMA helper: D = A(16x32 bf16) * B(32x16 bf16) + C(16x16 f32)
// ---------------------------------------------------------------------------
DEVFN v8f wmma_bf16(v16bf a, v16bf b, v8f c) {
  return __builtin_amdgcn_wmma_f32_16x16x32_bf16(
      /*neg_a=*/false, a, /*neg_b=*/false, b,
      /*c_mod=*/(short)0, c, /*reuse_a=*/false, /*reuse_b=*/false);
}

// A-fragment (16x32 bf16) from row-major [16 rows][stride].
// lane l: row = l%16; VGPR0..3 = K (l/16)*8..+7, VGPR4..7 = 16+(l/16)*8..+7.
DEVFN v16bf load_A16(const __bf16* base, int stride) {
  int l = threadIdx.x & 31;
  const __bf16* p = base + (l & 15) * stride + ((l >> 4) << 3);
  v8bf a = *(const v8bf*)p;
  v8bf b = *(const v8bf*)(p + 16);
  return __builtin_shufflevector(a, b, 0,1,2,3,4,5,6,7,8,9,10,11,12,13,14,15);
}

// B-fragment (32x16 bf16): memory buf[ncol][k]; lane l: ncol=l%16,
// VGPR0..3 = K (l/16)*8..+7, VGPR4..7 = 16+(l/16)*8..+7 -> two b128 loads.
DEVFN v16bf load_B16(const __bf16* base, int stride) {
  int l = threadIdx.x & 31;
  const __bf16* p = base + (l & 15) * stride + ((l >> 4) << 3);
  v8bf a = *(const v8bf*)p;
  v8bf b = *(const v8bf*)(p + 16);
  return __builtin_shufflevector(a, b, 0,1,2,3,4,5,6,7,8,9,10,11,12,13,14,15);
}

// ---------------------------------------------------------------------------
// Phase 0: fp32 -> bf16.  x[b][c][n] -> xT[b][n][c]; Wq/Wk/Wv -> bf16.
// ---------------------------------------------------------------------------
__global__ __launch_bounds__(256) void cvt_kernel(
    const float* __restrict__ x,
    const float* __restrict__ Wq, const float* __restrict__ Wk,
    const float* __restrict__ Wv,
    __bf16* __restrict__ xT,
    __bf16* __restrict__ Wqb, __bf16* __restrict__ Wkb,
    __bf16* __restrict__ Wvb) {
  int t = blockIdx.x * blockDim.x + threadIdx.x;
  if (t < BATCH * C * N) {
    int b = t / (C * N);
    int c = (t / N) % C;
    int n = t % N;
    xT[(size_t)b * N * C + (size_t)n * C + c] = (__bf16)x[t];
  }
  if (t < C * C) {
    Wqb[t] = (__bf16)Wq[t];
    Wkb[t] = (__bf16)Wk[t];
    Wvb[t] = (__bf16)Wv[t];
  }
}

// ---------------------------------------------------------------------------
// Phase 1: QKV projections, one 16x16 tile per wave (WMMA).
//   Qt[b][n][o] = sum_c xT[n][c]*Wq[o][c] + bq[o]   (same for Kt)
//   V [b][o][m] = sum_c Wv[o][c]*xT[m][c] + bv[o]
// ---------------------------------------------------------------------------
__global__ __launch_bounds__(256) void qkv_kernel(
    const __bf16* __restrict__ xT,
    const __bf16* __restrict__ Wqb, const __bf16* __restrict__ Wkb,
    const __bf16* __restrict__ Wvb,
    const float* __restrict__ bq, const float* __restrict__ bk,
    const float* __restrict__ bv,
    __bf16* __restrict__ Qt, __bf16* __restrict__ Kt,
    __bf16* __restrict__ Vn) {
  int wid  = blockIdx.x * (blockDim.x >> 5) + (threadIdx.x >> 5);
  int lane = threadIdx.x & 31;
  int mat  = wid % 3;
  int rest = wid / 3;
  int ot   = rest & 7;  rest >>= 3;
  int nt   = rest & 255;
  int b    = rest >> 8;

  const __bf16* xTb = xT + (size_t)b * N * C + (size_t)nt * 16 * C;
  v8f acc = {};

  if (mat < 2) {
    const __bf16* W = (mat == 0) ? Wqb : Wkb;
    #pragma unroll
    for (int cc = 0; cc < 4; ++cc) {
      v16bf a  = load_A16(xTb + cc * 32, C);
      v16bf bb = load_B16(W + ot * 16 * C + cc * 32, C);
      acc = wmma_bf16(a, bb, acc);
    }
    const float* bias = (mat == 0) ? bq : bk;
    float bval = bias[ot * 16 + (lane & 15)];
    __bf16* dst = ((mat == 0) ? Qt : Kt) + (size_t)b * N * C;
    int row0 = nt * 16 + ((lane >> 4) << 3);
    int col  = ot * 16 + (lane & 15);
    #pragma unroll
    for (int r = 0; r < 8; ++r)
      dst[(size_t)(row0 + r) * C + col] = (__bf16)(acc[r] + bval);
  } else {
    #pragma unroll
    for (int cc = 0; cc < 4; ++cc) {
      v16bf a  = load_A16(Wvb + ot * 16 * C + cc * 32, C);
      v16bf bb = load_B16(xTb + cc * 32, C);
      acc = wmma_bf16(a, bb, acc);
    }
    int orow0 = ot * 16 + ((lane >> 4) << 3);
    int col   = nt * 16 + (lane & 15);
    __bf16* dst = Vn + (size_t)b * C * N;
    #pragma unroll
    for (int r = 0; r < 8; ++r)
      dst[(size_t)(orow0 + r) * N + col] = (__bf16)(acc[r] + bv[orow0 + r]);
  }
}

// ---------------------------------------------------------------------------
// Phase 2: split-K flash attention with double-buffered async LDS staging.
// All 8 waves of a block share (g, b): K/V 64-key tiles are staged once per
// block into LDS (async, ASYNCcnt) and consumed by all waves via ds_load.
// Wave = (g, b, nt): 16 queries x 512 keys in 8 iterations of Bc=64,
// 32 WMMAs per iteration.  Emits partial O (fp32 16x128), m[16], l[16].
// ---------------------------------------------------------------------------
__global__ __launch_bounds__(256) void fa_kernel(
    const __bf16* __restrict__ Qt, const __bf16* __restrict__ Kt,
    const __bf16* __restrict__ Vn,
    float* __restrict__ Op, float* __restrict__ Mp, float* __restrict__ Lp) {
  __shared__ __align__(16) __bf16 Ks[2][64 * 128];   // [m_local][c]
  __shared__ __align__(16) __bf16 Vs[2][128 * 64];   // [c][m_local]
  __shared__ __align__(16) __bf16 Pbuf[8][16 * 64];  // per-wave P scratch

  int tid  = threadIdx.x;
  int wave = tid >> 5;
  int lane = tid & 31;
  int wid  = blockIdx.x * 8 + wave;   // 0..4095
  int nt   = wid & 255;               // per-wave query tile
  int rest = wid >> 8;                // block-uniform
  int b    = rest & 1;
  int g    = rest >> 1;

  const __bf16* Qb = Qt + (size_t)b * N * C + (size_t)nt * 16 * C;
  const __bf16* Kb = Kt + (size_t)b * N * C;  // block-uniform
  const __bf16* Vb = Vn + (size_t)b * C * N;  // block-uniform

  const int j0 = g * JPG;
  const int j1 = j0 + JPG;

  // Cooperative stage of one 64-key tile: K 64x128 (16KB) + V 128x64 (16KB);
  // 256 threads x 4 x two 16B async copies.
  auto stage = [&](int bi, int j) {
    const __bf16* gK = Kb + (size_t)j * 64 * C;
    const __bf16* gV = Vb + j * 64;
    #pragma unroll
    for (int i = 0; i < 4; ++i) {
      int flat = i * 256 + tid;                       // 0..1023 16B chunks
      async_cp16(gK + (flat >> 4) * C + (flat & 15) * 8,
                 &Ks[bi][(flat >> 4) * 128 + (flat & 15) * 8]);
      async_cp16(gV + (size_t)(flat >> 3) * N + (flat & 7) * 8,
                 &Vs[bi][(flat >> 3) * 64 + (flat & 7) * 8]);
    }
  };

  v16bf aq[4];
  #pragma unroll
  for (int cc = 0; cc < 4; ++cc) aq[cc] = load_A16(Qb + cc * 32, C);

  v8f   o_acc[8] = {};
  float mrow[8], lrow[8];
  #pragma unroll
  for (int r = 0; r < 8; ++r) { mrow[r] = -__builtin_inff(); lrow[r] = 0.0f; }

  __bf16* pw = &Pbuf[wave][0];
  const int rbase = (lane >> 4) << 3;
  const int col   = lane & 15;

  stage(0, j0);
  wait_async();
  __syncthreads();

  for (int j = j0; j < j1; ++j) {
    int bi = (j - j0) & 1;
    if (j + 1 < j1) stage(bi ^ 1, j + 1);   // block-uniform branch

    const __bf16* lK = &Ks[bi][0];
    const __bf16* lV = &Vs[bi][0];

    // S = Q K^T for 16x64 keys: four 16x16 D tiles, K-dim 128 in 4 chunks.
    v8f s[4] = {};
    #pragma unroll
    for (int cc = 0; cc < 4; ++cc) {
      #pragma unroll
      for (int mt = 0; mt < 4; ++mt) {
        v16bf bk = load_B16(lK + mt * 16 * 128 + cc * 32, 128);
        s[mt] = wmma_bf16(aq[cc], bk, s[mt]);
      }
    }

    // Online softmax across 64 keys (cols stripe lanes; rows stripe VGPRs).
    float scale[8];
    bool  upd = false;
    #pragma unroll
    for (int r = 0; r < 8; ++r) {
      float t = fmaxf(fmaxf(s[0][r], s[1][r]), fmaxf(s[2][r], s[3][r]));
      t = fmaxf(t, __shfl_xor(t, 1));
      t = fmaxf(t, __shfl_xor(t, 2));
      t = fmaxf(t, __shfl_xor(t, 4));
      t = fmaxf(t, __shfl_xor(t, 8));
      float mold = mrow[r];
      float mnew = fmaxf(mold, t);
      upd = upd || (mnew > mold);
      float sc = __expf(mold - mnew);
      float p0 = __expf(s[0][r] - mnew);
      float p1 = __expf(s[1][r] - mnew);
      float p2 = __expf(s[2][r] - mnew);
      float p3 = __expf(s[3][r] - mnew);
      s[0][r] = p0; s[1][r] = p1; s[2][r] = p2; s[3][r] = p3;
      float rs = (p0 + p1) + (p2 + p3);
      rs += __shfl_xor(rs, 1);
      rs += __shfl_xor(rs, 2);
      rs += __shfl_xor(rs, 4);
      rs += __shfl_xor(rs, 8);
      lrow[r]  = lrow[r] * sc + rs;
      mrow[r]  = mnew;
      scale[r] = sc;
    }
    if (__ballot(upd) != 0) {  // wave-uniform; EXEC untouched
      #pragma unroll
      for (int cf = 0; cf < 8; ++cf)
        #pragma unroll
        for (int r = 0; r < 8; ++r) o_acc[cf][r] *= scale[r];
    }

    // P (16x64) -> bf16 A-fragments via wave-private LDS (in-order DS).
    #pragma unroll
    for (int mt = 0; mt < 4; ++mt)
      #pragma unroll
      for (int r = 0; r < 8; ++r)
        pw[(rbase + r) * 64 + mt * 16 + col] = (__bf16)s[mt][r];
    v16bf ap0 = load_A16(pw, 64);        // keys j*64    .. +31
    v16bf ap1 = load_A16(pw + 32, 64);   // keys j*64+32 .. +63

    // O += P * V^T : 8 channel tiles x 2 key halves, from staged LDS V.
    #pragma unroll
    for (int cf = 0; cf < 8; ++cf) {
      v16bf bv0 = load_B16(lV + cf * 16 * 64, 64);
      v16bf bv1 = load_B16(lV + cf * 16 * 64 + 32, 64);
      o_acc[cf] = wmma_bf16(ap0, bv0, o_acc[cf]);
      o_acc[cf] = wmma_bf16(ap1, bv1, o_acc[cf]);
    }

    // Hand off double buffer: own async writes done, then block-wide sync.
    wait_async();
    __syncthreads();
  }

  // Store partials: O (row-major [16][128] f32), m[16], l[16].
  size_t tile = ((size_t)g * BATCH + b) * NT + nt;
  float* Ob = Op + tile * (16 * C);
  #pragma unroll
  for (int cf = 0; cf < 8; ++cf)
    #pragma unroll
    for (int r = 0; r < 8; ++r)
      Ob[(size_t)(rbase + r) * C + cf * 16 + col] = o_acc[cf][r];
  if (col == 0) {  // lanes 0 and 16 cover rows 0..7 and 8..15
    #pragma unroll
    for (int r = 0; r < 8; ++r) {
      Mp[tile * 16 + rbase + r] = mrow[r];
      Lp[tile * 16 + rbase + r] = lrow[r];
    }
  }
}

// ---------------------------------------------------------------------------
// Phase 3: combine split-K partials.  One block per (b, nt).
//   out[b][c][n] = sum_g w_g O_g[n][c],  w_g = exp(m_g - M) / sum l_g e^...
// ---------------------------------------------------------------------------
__global__ __launch_bounds__(256) void combine_kernel(
    const float* __restrict__ Op, const float* __restrict__ Mp,
    const float* __restrict__ Lp, float* __restrict__ out) {
  __shared__ float sM[16], sLinv[16], sW[G][16];
  int tile = blockIdx.x;            // 0..511 = b*NT + nt
  int b  = tile >> 8;
  int nt = tile & 255;
  int tid = threadIdx.x;

  if (tid < 16) {
    float M = -__builtin_inff();
    #pragma unroll
    for (int g = 0; g < G; ++g)
      M = fmaxf(M, Mp[(((size_t)g * BATCH + b) * NT + nt) * 16 + tid]);
    float L = 0.0f;
    #pragma unroll
    for (int g = 0; g < G; ++g) {
      size_t idx = (((size_t)g * BATCH + b) * NT + nt) * 16 + tid;
      L += Lp[idx] * __expf(Mp[idx] - M);
    }
    sM[tid] = M;
    sLinv[tid] = 1.0f / L;
  }
  __syncthreads();
  if (tid < G * 16) {
    int g = tid >> 4, r = tid & 15;
    size_t idx = (((size_t)g * BATCH + b) * NT + nt) * 16 + r;
    sW[g][r] = __expf(Mp[idx] - sM[r]) * sLinv[r];
  }
  __syncthreads();

  #pragma unroll
  for (int k = 0; k < 8; ++k) {
    int e = tid + k * 256;          // 0..2047 over [n=16][c=128]
    int n = e >> 7, c = e & 127;
    float acc = 0.0f;
    #pragma unroll
    for (int g = 0; g < G; ++g)
      acc += sW[g][n] * Op[(((size_t)g * BATCH + b) * NT + nt) * (16 * C) + e];
    out[((size_t)b * C + c) * N + nt * 16 + n] = acc;
  }
}

// ---------------------------------------------------------------------------
// Launch
// ---------------------------------------------------------------------------
extern "C" void kernel_launch(void* const* d_in, const int* in_sizes, int n_in,
                              void* d_out, int out_size, void* d_ws,
                              size_t ws_size, hipStream_t stream) {
  (void)in_sizes; (void)n_in; (void)out_size; (void)ws_size;
  const float* x  = (const float*)d_in[0];
  const float* Wq = (const float*)d_in[1];
  const float* bq = (const float*)d_in[2];
  const float* Wk = (const float*)d_in[3];
  const float* bk = (const float*)d_in[4];
  const float* Wv = (const float*)d_in[5];
  const float* bv = (const float*)d_in[6];
  float* out = (float*)d_out;

  char* ws = (char*)d_ws;
  size_t off = 0;
  __bf16* xT  = (__bf16*)(ws + off); off += (size_t)BATCH * N * C * 2;
  __bf16* Qt  = (__bf16*)(ws + off); off += (size_t)BATCH * N * C * 2;
  __bf16* Kt  = (__bf16*)(ws + off); off += (size_t)BATCH * N * C * 2;
  __bf16* Vn  = (__bf16*)(ws + off); off += (size_t)BATCH * C * N * 2;
  __bf16* Wqb = (__bf16*)(ws + off); off += (size_t)C * C * 2;
  __bf16* Wkb = (__bf16*)(ws + off); off += (size_t)C * C * 2;
  __bf16* Wvb = (__bf16*)(ws + off); off += (size_t)C * C * 2;
  off = (off + 255) & ~(size_t)255;
  float* Op = (float*)(ws + off); off += (size_t)G * BATCH * NT * 16 * C * 4; // 32MB
  float* Mp = (float*)(ws + off); off += (size_t)G * BATCH * NT * 16 * 4;
  float* Lp = (float*)(ws + off); off += (size_t)G * BATCH * NT * 16 * 4;

  int total = BATCH * C * N;
  cvt_kernel<<<(total + 255) / 256, 256, 0, stream>>>(x, Wq, Wk, Wv, xT, Wqb,
                                                      Wkb, Wvb);

  // QKV: 2 * 256 * 8 * 3 = 12288 waves.
  qkv_kernel<<<12288 / 8, 256, 0, stream>>>(xT, Wqb, Wkb, Wvb, bq, bk, bv, Qt,
                                            Kt, Vn);

  // Split-K flash attention: 8 * 2 * 256 = 4096 waves, 512 blocks.
  fa_kernel<<<4096 / 8, 256, 0, stream>>>(Qt, Kt, Vn, Op, Mp, Lp);

  // Combine: one block per (b, nt).
  combine_kernel<<<BATCH * NT, 256, 0, stream>>>(Op, Mp, Lp, out);
}
